// LocalEncoder_14800457302123
// MI455X (gfx1250) — matrix operand order repaired and checked by reading, not compiled, not run on hardware
//
#include <hip/hip_runtime.h>

// ---------------------------------------------------------------------------
// BLT local encoder for MI455X (gfx1250, wave32, WMMA).
// Residual stream kept in fp32 in d_out; all GEMMs run bf16 x bf16 -> fp32
// via v_wmma_f32_16x16x32_bf16. Entropy predictor kept fp32 (threshold-
// sensitive). Attention is flash-style with exact block-diagonal masking.
// GEMM B tiles and attention V tiles are staged through LDS; the B-tile copy
// uses gfx1250 async global->LDS loads when the toolchain exposes them.
// ---------------------------------------------------------------------------

typedef __attribute__((ext_vector_type(16))) __bf16 v16bf;
typedef __attribute__((ext_vector_type(8)))  __bf16 v8bf;
typedef __attribute__((ext_vector_type(8)))  float  v8f;
typedef int v4i_vs __attribute__((__vector_size__(16)));  // async-copy unit

static constexpr int Bc  = 4;
static constexpr int Sc  = 1024;
static constexpr int Hc  = 512;
static constexpr int NHc = 8;
static constexpr int HDc = 64;
static constexpr int FFc = 2048;
static constexpr int Lc  = 4;
static constexpr int Vc  = 256;
static constexpr int BSc = Bc * Sc;   // 4096 rows
static constexpr int H3c = 3 * Hc;    // 1536
static constexpr int LDP = 48;        // padded LDS row stride (bf16 elems)

#if defined(__gfx1250__) &&                                                   \
    __has_builtin(__builtin_amdgcn_global_load_async_to_lds_b128) &&          \
    __has_builtin(__builtin_amdgcn_s_wait_asynccnt)
#define USE_ASYNC_LDS 1
#else
#define USE_ASYNC_LDS 0
#endif

__device__ __forceinline__ v8f wmma_bf16(v16bf a, v16bf b, v8f c) {
  return __builtin_amdgcn_wmma_f32_16x16x32_bf16(
      false, a, false, b, (short)0, c, false, false);
}

// A fragment (16x32, bf16): lane holds row m = lane&15.
// element j -> k = (lane>>4)*16 + j. Loaded as two 16B chunks.
__device__ __forceinline__ v16bf load_a16x32(const __bf16* X, int ld, int m0,
                                             int k0, int lane) {
  int m  = m0 + (lane & 15);
  int kb = k0 + ((lane >> 4) << 4);
  const __bf16* p = X + (size_t)m * ld + kb;
  v8bf lo = *(const v8bf*)p;
  v8bf hi = *(const v8bf*)(p + 8);
  v16bf r;
#pragma unroll
  for (int i = 0; i < 8; ++i) { r[i] = lo[i]; r[8 + i] = hi[i]; }
  return r;
}

// B fragment (32x16, bf16) from a row-major [N,K]-style matrix with row
// stride ld (B[k][n] = W[n][k]): lane holds column n = lane&15;
// element j -> k = (lane>>4)*16 + j (one contiguous 32B read).
__device__ __forceinline__ v16bf load_b32x16(const __bf16* W, int ld, int n0,
                                             int k0, int lane) {
  int n  = n0 + (lane & 15);
  int kb = k0 + ((lane >> 4) << 4);
  return *(const v16bf*)(W + (size_t)n * ld + kb);
}

// Cooperative copy of a 64(N) x 32(K) bf16 tile from global [N,K] (row
// stride K) into padded LDS [64][LDP]. 128 threads, 32B per thread.
__device__ __forceinline__ void stage_B(const __bf16* __restrict__ W, int K,
                                        int n0, int k0, __bf16 (*Bs)[LDP],
                                        int tid) {
  int r  = tid >> 1;            // 0..63
  int kc = (tid & 1) << 4;      // 0 or 16
  const __bf16* g = W + (size_t)(n0 + r) * K + k0 + kc;
#if USE_ASYNC_LDS
#pragma unroll
  for (int j = 0; j < 2; ++j) {
    __builtin_amdgcn_global_load_async_to_lds_b128(
        (__attribute__((address_space(1))) v4i_vs*)(__bf16*)(g + j * 8),
        (__attribute__((address_space(3))) v4i_vs*)&Bs[r][kc + j * 8], 0, 0);
  }
#else
  v8bf t0 = *(const v8bf*)g;
  v8bf t1 = *(const v8bf*)(g + 8);
  *(v8bf*)&Bs[r][kc]     = t0;
  *(v8bf*)&Bs[r][kc + 8] = t1;
#endif
}

// ---------------------------------------------------------------------------
// Elementwise helpers
// ---------------------------------------------------------------------------
__global__ void f2bf_kernel(const float* __restrict__ in,
                            __bf16* __restrict__ out, int n) {
  for (int i = blockIdx.x * blockDim.x + threadIdx.x; i < n;
       i += gridDim.x * blockDim.x)
    out[i] = (__bf16)in[i];
}

__global__ void embed_kernel(const int* __restrict__ bytes,
                             const float* __restrict__ emb,
                             float* __restrict__ x, int total) {
  for (int i = blockIdx.x * blockDim.x + threadIdx.x; i < total;
       i += gridDim.x * blockDim.x) {
    int row = i >> 9;          // H = 512
    int e   = i & 511;
    x[i] = emb[(size_t)bytes[row] * Hc + e];
  }
}

// ---------------------------------------------------------------------------
// Entropy predictor (fp32): one block per sequence position, 256 threads = V.
// ---------------------------------------------------------------------------
__global__ __launch_bounds__(256) void entropy_kernel(
    const int* __restrict__ bytes, const float* __restrict__ pred_emb,
    const float* __restrict__ pred_w, const float* __restrict__ pred_b,
    int* __restrict__ boundary) {
  __shared__ float se[Hc];
  __shared__ float red[Vc];
  int s = blockIdx.x, tid = threadIdx.x;
  int token = bytes[s];
  for (int i = tid; i < Hc; i += 256) se[i] = pred_emb[(size_t)token * Hc + i];
  __syncthreads();
  float logit = pred_b[tid];
  for (int h = 0; h < Hc; ++h) logit += se[h] * pred_w[(size_t)h * Vc + tid];
  // max
  red[tid] = logit; __syncthreads();
  for (int off = 128; off > 0; off >>= 1) {
    if (tid < off) red[tid] = fmaxf(red[tid], red[tid + off]);
    __syncthreads();
  }
  float mx = red[0]; __syncthreads();
  float e = expf(logit - mx);
  red[tid] = e; __syncthreads();
  for (int off = 128; off > 0; off >>= 1) {
    if (tid < off) red[tid] += red[tid + off];
    __syncthreads();
  }
  float Z = red[0]; __syncthreads();
  float pr = e / Z;
  red[tid] = pr * log2f(pr + 1e-9f); __syncthreads();
  for (int off = 128; off > 0; off >>= 1) {
    if (tid < off) red[tid] += red[tid + off];
    __syncthreads();
  }
  if (tid == 0) boundary[s] = (-red[0] > 0.8f) ? 1 : 0;
}

// seg[i] = sum_{j<i} boundary[j]  (single block, 1024 threads)
__global__ __launch_bounds__(1024) void seg_scan_kernel(
    const int* __restrict__ boundary, int* __restrict__ seg) {
  __shared__ int tmp[Sc];
  int tid = threadIdx.x;
  tmp[tid] = (tid > 0) ? boundary[tid - 1] : 0;
  __syncthreads();
  for (int off = 1; off < Sc; off <<= 1) {
    int v = (tid >= off) ? tmp[tid - off] : 0;
    __syncthreads();
    tmp[tid] += v;
    __syncthreads();
  }
  seg[tid] = tmp[tid];
}

// ---------------------------------------------------------------------------
// LayerNorm (fp32 in, bf16 out). One block (128 thr) per row, H = 512.
// ---------------------------------------------------------------------------
__global__ __launch_bounds__(128) void ln_kernel(
    const float* __restrict__ x, const float* __restrict__ w,
    const float* __restrict__ b, __bf16* __restrict__ out) {
  __shared__ float r1[128], r2[128];
  int row = blockIdx.x, tid = threadIdx.x;
  const float* xr = x + (size_t)row * Hc;
  float v[4], s = 0.f, s2 = 0.f;
#pragma unroll
  for (int i = 0; i < 4; ++i) {
    v[i] = xr[tid + i * 128];
    s += v[i]; s2 += v[i] * v[i];
  }
  r1[tid] = s; r2[tid] = s2; __syncthreads();
  for (int off = 64; off > 0; off >>= 1) {
    if (tid < off) { r1[tid] += r1[tid + off]; r2[tid] += r2[tid + off]; }
    __syncthreads();
  }
  float mean = r1[0] * (1.f / Hc);
  float var  = r2[0] * (1.f / Hc) - mean * mean;
  float rs   = rsqrtf(var + 1e-5f);
#pragma unroll
  for (int i = 0; i < 4; ++i) {
    int e = tid + i * 128;
    out[(size_t)row * Hc + e] = (__bf16)((v[i] - mean) * rs * w[e] + b[e]);
  }
}

// ---------------------------------------------------------------------------
// WMMA GEMM: C[M,N] = A[M,K] (bf16) x W[N,K]^T (bf16) + bias, fused epilogues.
// Wave tile 16x64, 4 waves/block -> 64x64 block tile. B tile (shared by all
// 4 waves) staged double-buffered through LDS (async copy when available);
// A rows are wave-private and stream straight from global.
// mode 0: fp32 store (+optional residual, +optional bf16 copy)
// mode 1: ReLU, bf16 store to Cb
// mode 2: QKV scatter to per-head bf16 buffers [B*NH, S, HD]
// ---------------------------------------------------------------------------
__global__ __launch_bounds__(128) void gemm_bf16_kernel(
    const __bf16* __restrict__ A, const __bf16* __restrict__ W,
    const float* __restrict__ bias, const float* __restrict__ resid,
    float* __restrict__ C, __bf16* __restrict__ Cb,
    __bf16* __restrict__ qb, __bf16* __restrict__ kb, __bf16* __restrict__ vb,
    int M, int N, int K, int mode) {
  __shared__ __bf16 Bs[2][64][LDP];

  int tid = threadIdx.x, lane = tid & 31, wid = tid >> 5;
  int m0 = blockIdx.y * 64 + wid * 16;
  int n0 = blockIdx.x * 64;

  v8f acc[4];
#pragma unroll
  for (int t = 0; t < 4; ++t)
#pragma unroll
    for (int i = 0; i < 8; ++i) acc[t][i] = 0.f;

  stage_B(W, K, n0, 0, Bs[0], tid);
  int buf = 0;
  for (int k0 = 0; k0 < K; k0 += 32) {
    if (k0 + 32 < K) {
      stage_B(W, K, n0, k0 + 32, Bs[buf ^ 1], tid);
#if USE_ASYNC_LDS
      __builtin_amdgcn_s_wait_asynccnt(2);  // current buf done; prefetch flies
#endif
    } else {
#if USE_ASYNC_LDS
      __builtin_amdgcn_s_wait_asynccnt(0);
#endif
    }
    __syncthreads();

    v16bf a = load_a16x32(A, K, m0, k0, lane);
#pragma unroll
    for (int t = 0; t < 4; ++t) {
      v16bf bf = load_b32x16(&Bs[buf][0][0], LDP, t * 16, 0, lane);
      acc[t] = wmma_bf16(a, bf, acc[t]);
    }
    __syncthreads();
    buf ^= 1;
  }

  int rbase = m0 + ((lane >> 4) << 3);
  int cb0   = n0 + (lane & 15);
#pragma unroll
  for (int t = 0; t < 4; ++t) {
    int col = cb0 + t * 16;
    float bv = bias ? bias[col] : 0.f;
#pragma unroll
    for (int i = 0; i < 8; ++i) {
      int row = rbase + i;
      float v = acc[t][i] + bv;
      if (resid) v += resid[(size_t)row * N + col];
      if (mode == 1) v = fmaxf(v, 0.f);
      if (mode == 2) {
        int bidx = row >> 10, srow = row & (Sc - 1);
        int t3 = col / Hc, e = col - t3 * Hc;
        int hh = e >> 6, d = e & (HDc - 1);
        __bf16* dst = (t3 == 0) ? qb : ((t3 == 1) ? kb : vb);
        dst[((((size_t)bidx * NHc + hh) * Sc) + srow) * HDc + d] = (__bf16)v;
      } else {
        if (Cb) Cb[(size_t)row * N + col] = (__bf16)v;
        if (C)  C [(size_t)row * N + col] = v;
      }
    }
  }
}

// ---------------------------------------------------------------------------
// Flash attention with block-diagonal segment mask.
// Each wave: 16 queries x HD=64; 4 waves/block share one (b,h) 64-query span.
// V tiles staged transposed through LDS so P x V B-fragments are one 32B read.
// ---------------------------------------------------------------------------
__global__ __launch_bounds__(128) void attn_kernel(
    const __bf16* __restrict__ qm, const __bf16* __restrict__ km,
    const __bf16* __restrict__ vm, const int* __restrict__ seg,
    float* __restrict__ o) {
  __shared__ int    ssg[Sc];
  __shared__ __bf16 sp[4][16][32];
  __shared__ __bf16 Vt[64][LDP];     // Vt[n][k]: transposed 32x64 V tile

  int tid = threadIdx.x, lane = tid & 31, wid = tid >> 5;
  for (int i = tid; i < Sc; i += 128) ssg[i] = seg[i];
  __syncthreads();

  const int nqb = Sc / 64;                  // 16 query-blocks per (b,h)
  int bh  = blockIdx.x / nqb;
  int q0b = (blockIdx.x % nqb) * 64;
  int q0  = q0b + wid * 16;
  int b   = bh >> 3, h = bh & 7;

  const __bf16* Q  = qm + (size_t)bh * Sc * HDc;
  const __bf16* Kp = km + (size_t)bh * Sc * HDc;
  const __bf16* Vp = vm + (size_t)bh * Sc * HDc;

  v16bf aq0 = load_a16x32(Q, HDc, q0, 0, lane);
  v16bf aq1 = load_a16x32(Q, HDc, q0, 32, lane);

  int sq[8];
#pragma unroll
  for (int i = 0; i < 8; ++i) sq[i] = ssg[q0 + ((lane >> 4) << 3) + i];
  int segqlo = ssg[q0b], segqhi = ssg[q0b + 63];

  float mi[8], li[8];
  v8f acc[4];
#pragma unroll
  for (int i = 0; i < 8; ++i) { mi[i] = -1e30f; li[i] = 0.f; }
#pragma unroll
  for (int t = 0; t < 4; ++t)
#pragma unroll
    for (int i = 0; i < 8; ++i) acc[t][i] = 0.f;

  for (int k0 = 0; k0 < Sc; k0 += 32) {
    // block-uniform segment-range pruning (seg is non-decreasing)
    if (ssg[k0] > segqhi) break;
    if (ssg[k0 + 31] < segqlo) continue;

    // cooperative transposed stage of V tile: Vt[n][k] = V[k0+k][n]
    {
      int kr = tid >> 2;              // 0..31
      int cb = (tid & 3) << 4;        // 0,16,32,48
      v16bf vv = *(const v16bf*)(Vp + (size_t)(k0 + kr) * HDc + cb);
#pragma unroll
      for (int j = 0; j < 16; ++j) Vt[cb + j][kr] = vv[j];
    }

    v8f sc0, sc1;
#pragma unroll
    for (int i = 0; i < 8; ++i) { sc0[i] = 0.f; sc1[i] = 0.f; }
    {
      v16bf b0 = load_b32x16(Kp, HDc, k0,       0,  lane);
      v16bf b1 = load_b32x16(Kp, HDc, k0,       32, lane);
      sc0 = wmma_bf16(aq0, b0, sc0);
      sc0 = wmma_bf16(aq1, b1, sc0);
      v16bf b2 = load_b32x16(Kp, HDc, k0 + 16,  0,  lane);
      v16bf b3 = load_b32x16(Kp, HDc, k0 + 16,  32, lane);
      sc1 = wmma_bf16(aq0, b2, sc1);
      sc1 = wmma_bf16(aq1, b3, sc1);
    }

    int sk0 = ssg[k0 + (lane & 15)];
    int sk1 = ssg[k0 + 16 + (lane & 15)];

    float p0[8], p1[8], alpha[8];
#pragma unroll
    for (int i = 0; i < 8; ++i) {
      float s0 = (sq[i] == sk0) ? sc0[i] * 0.125f : -1e30f;
      float s1 = (sq[i] == sk1) ? sc1[i] * 0.125f : -1e30f;
      float rm = fmaxf(s0, s1);
#pragma unroll
      for (int msk = 1; msk < 16; msk <<= 1)
        rm = fmaxf(rm, __shfl_xor(rm, msk, 32));
      float mnew = fmaxf(mi[i], rm);
      float e0 = (s0 <= -1e29f) ? 0.f : __expf(s0 - mnew);
      float e1 = (s1 <= -1e29f) ? 0.f : __expf(s1 - mnew);
      float rs = e0 + e1;
#pragma unroll
      for (int msk = 1; msk < 16; msk <<= 1)
        rs += __shfl_xor(rs, msk, 32);
      alpha[i] = __expf(mi[i] - mnew);
      li[i] = li[i] * alpha[i] + rs;
      mi[i] = mnew;
      p0[i] = e0; p1[i] = e1;
    }
#pragma unroll
    for (int t = 0; t < 4; ++t)
#pragma unroll
      for (int i = 0; i < 8; ++i) acc[t][i] *= alpha[i];

    // stage P (16x32) in LDS, re-read in A-fragment layout
    int rr = ((lane >> 4) << 3);
#pragma unroll
    for (int i = 0; i < 8; ++i) {
      sp[wid][rr + i][lane & 15]        = (__bf16)p0[i];
      sp[wid][rr + i][16 + (lane & 15)] = (__bf16)p1[i];
    }
    __syncthreads();                 // sp + Vt visible block-wide
    v16bf ap;
    {
      int m = lane & 15, kb2 = ((lane >> 4) << 4);
#pragma unroll
      for (int j = 0; j < 8; ++j) {
        ap[j]     = sp[wid][m][kb2 + j];
        ap[8 + j] = sp[wid][m][kb2 + 8 + j];
      }
    }

    // P (16x32) x V (32x64): 4 WMMAs, B fragments straight from Vt
#pragma unroll
    for (int t = 0; t < 4; ++t) {
      v16bf bv = *(const v16bf*)&Vt[t * 16 + (lane & 15)][(lane >> 4) << 4];
      acc[t] = wmma_bf16(ap, bv, acc[t]);
    }
    __syncthreads();                 // protect sp/Vt before next overwrite
  }

#pragma unroll
  for (int t = 0; t < 4; ++t)
#pragma unroll
    for (int i = 0; i < 8; ++i) {
      int srow = q0 + ((lane >> 4) << 3) + i;
      float denom = li[i];
      float v = (denom > 0.f) ? acc[t][i] / denom : 0.f;
      o[(((size_t)b * Sc) + srow) * Hc + h * HDc + t * 16 + (lane & 15)] = v;
    }
}

// ---------------------------------------------------------------------------
// Orchestration
// ---------------------------------------------------------------------------
extern "C" void kernel_launch(void* const* d_in, const int* in_sizes, int n_in,
                              void* d_out, int out_size, void* d_ws,
                              size_t ws_size, hipStream_t stream) {
  (void)in_sizes; (void)n_in; (void)out_size; (void)ws_size;

  const int*   bytes    = (const int*)  d_in[0];
  const float* emb      = (const float*)d_in[1];
  const float* pred_emb = (const float*)d_in[2];
  const float* pred_w   = (const float*)d_in[3];
  const float* pred_b   = (const float*)d_in[4];
  const float* ln1_w    = (const float*)d_in[5];
  const float* ln1_b    = (const float*)d_in[6];
  const float* ln2_w    = (const float*)d_in[7];
  const float* ln2_b    = (const float*)d_in[8];
  const float* qkv_w    = (const float*)d_in[9];
  const float* qkv_b    = (const float*)d_in[10];
  const float* out_w    = (const float*)d_in[11];
  const float* out_b    = (const float*)d_in[12];
  const float* ff1_w    = (const float*)d_in[13];
  const float* ff1_b    = (const float*)d_in[14];
  const float* ff2_w    = (const float*)d_in[15];
  const float* ff2_b    = (const float*)d_in[16];

  float* x = (float*)d_out;   // residual stream lives in d_out (B*S*H fp32)

  char* p = (char*)d_ws;
  auto alloc = [&](size_t nbytes) -> char* {
    char* r = p;
    p += (nbytes + 255) & ~(size_t)255;
    return r;
  };

  const size_t nWq  = (size_t)Lc * H3c * Hc;
  const size_t nWo  = (size_t)Lc * Hc  * Hc;
  const size_t nWf1 = (size_t)Lc * FFc * Hc;
  const size_t nWf2 = (size_t)Lc * Hc  * FFc;

  __bf16* wq   = (__bf16*)alloc(sizeof(__bf16) * nWq);
  __bf16* wo   = (__bf16*)alloc(sizeof(__bf16) * nWo);
  __bf16* wf1  = (__bf16*)alloc(sizeof(__bf16) * nWf1);
  __bf16* wf2  = (__bf16*)alloc(sizeof(__bf16) * nWf2);
  __bf16* hb   = (__bf16*)alloc(sizeof(__bf16) * (size_t)BSc * Hc);
  __bf16* fbuf = (__bf16*)alloc(sizeof(__bf16) * (size_t)BSc * FFc);
  __bf16* qbuf = (__bf16*)alloc(sizeof(__bf16) * (size_t)BSc * Hc);
  __bf16* kbuf = (__bf16*)alloc(sizeof(__bf16) * (size_t)BSc * Hc);
  __bf16* vbuf = (__bf16*)alloc(sizeof(__bf16) * (size_t)BSc * Hc);
  float*  obuf = (float*) alloc(sizeof(float)  * (size_t)BSc * Hc);
  int* boundary = (int*)alloc(sizeof(int) * Sc);
  int* seg      = (int*)alloc(sizeof(int) * Sc);

  auto cvt = [&](const float* src, __bf16* dst, size_t n) {
    int blocks = (int)((n + 255) / 256);
    f2bf_kernel<<<blocks, 256, 0, stream>>>(src, dst, (int)n);
  };

  // 1) weights -> bf16
  cvt(qkv_w, wq,  nWq);
  cvt(out_w, wo,  nWo);
  cvt(ff1_w, wf1, nWf1);
  cvt(ff2_w, wf2, nWf2);

  // 2) embedding -> residual stream
  {
    int total = BSc * Hc;
    embed_kernel<<<(total + 255) / 256, 256, 0, stream>>>(bytes, emb, x, total);
  }

  // 3) entropy boundaries (fp32) + segment scan
  entropy_kernel<<<Sc, 256, 0, stream>>>(bytes, pred_emb, pred_w, pred_b,
                                         boundary);
  seg_scan_kernel<<<1, Sc, 0, stream>>>(boundary, seg);

  // 4) encoder layers
  for (int l = 0; l < Lc; ++l) {
    // LN1 -> hb (bf16)
    ln_kernel<<<BSc, 128, 0, stream>>>(x, ln1_w + (size_t)l * Hc,
                                       ln1_b + (size_t)l * Hc, hb);
    // QKV GEMM with per-head bf16 scatter
    gemm_bf16_kernel<<<dim3(H3c / 64, BSc / 64), 128, 0, stream>>>(
        hb, wq + (size_t)l * H3c * Hc, qkv_b + (size_t)l * H3c,
        nullptr, nullptr, nullptr, qbuf, kbuf, vbuf, BSc, H3c, Hc, 2);
    // attention
    attn_kernel<<<Bc * NHc * (Sc / 64), 128, 0, stream>>>(qbuf, kbuf, vbuf,
                                                          seg, obuf);
    // attn out -> bf16, out-proj with residual add into x
    cvt(obuf, hb, (size_t)BSc * Hc);
    gemm_bf16_kernel<<<dim3(Hc / 64, BSc / 64), 128, 0, stream>>>(
        hb, wo + (size_t)l * Hc * Hc, out_b + (size_t)l * Hc,
        x, x, nullptr, nullptr, nullptr, nullptr, BSc, Hc, Hc, 0);
    // LN2 -> hb
    ln_kernel<<<BSc, 128, 0, stream>>>(x, ln2_w + (size_t)l * Hc,
                                       ln2_b + (size_t)l * Hc, hb);
    // FF1 (ReLU, bf16 out)
    gemm_bf16_kernel<<<dim3(FFc / 64, BSc / 64), 128, 0, stream>>>(
        hb, wf1 + (size_t)l * FFc * Hc, ff1_b + (size_t)l * FFc,
        nullptr, nullptr, fbuf, nullptr, nullptr, nullptr, BSc, FFc, Hc, 1);
    // FF2 with residual add into x
    gemm_bf16_kernel<<<dim3(Hc / 64, BSc / 64), 128, 0, stream>>>(
        fbuf, wf2 + (size_t)l * Hc * FFc, ff2_b + (size_t)l * Hc,
        x, x, nullptr, nullptr, nullptr, nullptr, BSc, Hc, FFc, 0);
  }
}
